// SimAMCNNMogrifierLSTM_67018669687406
// MI455X (gfx1250) — matrix operand order repaired and checked by reading, not compiled, not run on hardware
//
#include <hip/hip_runtime.h>

#define B_  256
#define T_  64
#define D_  128
#define C1_ 32
#define C2_ 64
#define H_  1024
#define F_  (C2_ * D_)     // 8192
#define NC_ 5
#define G4H (4 * H_)       // 4096

typedef __attribute__((ext_vector_type(16))) __bf16 bf16x16;
typedef __attribute__((ext_vector_type(8)))  float  f32x8;
typedef __attribute__((ext_vector_type(4)))  unsigned int u32x4;
typedef __attribute__((ext_vector_type(8)))  int  i32x8;
typedef __attribute__((ext_vector_type(4)))  int  i32x4;

__device__ __forceinline__ float sigmoidf_(float x) { return 1.0f / (1.0f + __expf(-x)); }

// ---------------------------------------------------------------------------
// Block-wide sum reduction (wave32 shuffle + LDS), 256 threads = 8 waves.
// ---------------------------------------------------------------------------
__device__ __forceinline__ float block_reduce_sum(float v, float* sbuf) {
  __syncthreads();                       // protect sbuf reuse across calls
#pragma unroll
  for (int off = 16; off > 0; off >>= 1) v += __shfl_xor(v, off, 32);
  const int lane = threadIdx.x & 31;
  const int w    = threadIdx.x >> 5;
  if (lane == 0) sbuf[w] = v;
  __syncthreads();
  if (threadIdx.x == 0) {
    float s = 0.f;
#pragma unroll
    for (int i = 0; i < 8; ++i) s += sbuf[i];
    sbuf[15] = s;
  }
  __syncthreads();
  return sbuf[15];
}

// ---------------------------------------------------------------------------
// Tensor Data Mover: DMA a 2D tile (tile_d0 x tile_d1 elements, 2B each) from
// global memory into LDS at byte offset lds_byte_off, with LDS padding of
// 4 DWORDs (16B = 8 bf16) after every 16 DWORDs (64B = one 32-elem row).
// D# built per CDNA5 ISA 8.3/8.4 (group0: count/lds/global/type=2; group1:
// data_size=2B, pad_enable, dims, strides; groups 2/3 zero for 2D tiles).
// This toolchain's builtin takes 6 args (g0, g1, g2, g3, g4, cpol).
// ---------------------------------------------------------------------------
__device__ __forceinline__ void tdm_load_2d(unsigned lds_byte_off, const void* gaddr,
                                            unsigned tensor_d0, unsigned tensor_d1,
                                            unsigned tile_d0, unsigned tile_d1,
                                            unsigned long long stride0_elems)
{
  const unsigned long long ga = (unsigned long long)(uintptr_t)gaddr;
  u32x4 g0;
  g0.x = 1u;                                               // count=1: valid user D#
  g0.y = lds_byte_off;                                     // lds_addr (bytes)
  g0.z = (unsigned)ga;                                     // global_addr[31:0]
  g0.w = (unsigned)((ga >> 32) & 0x1FFFFFFu) | (2u << 30); // global_addr[56:32] | type=2
  i32x8 g1;
  g1[0] = (int)((1u << 16)      // data_size = 2 bytes
              | (1u << 20)      // pad_enable
              | (3u << 22)      // pad_interval = 16 DWORDs (one 64B row)
              | (3u << 25));    // pad_amount   = 4 DWORDs  (16B row pad)
  g1[1] = (int)((tensor_d0 & 0xFFFFu) << 16);                          // tdim0[15:0]
  g1[2] = (int)((tensor_d0 >> 16) | ((tensor_d1 & 0xFFFFu) << 16));    // tdim0[31:16]|tdim1[15:0]
  g1[3] = (int)((tensor_d1 >> 16) | ((tile_d0 & 0xFFFFu) << 16));      // tdim1[31:16]|tile0
  g1[4] = (int)(tile_d1 & 0xFFFFu);                                    // tile1 (tile2=0)
  g1[5] = (int)(unsigned)(stride0_elems & 0xFFFFFFFFull);              // dim0_stride[31:0]
  g1[6] = (int)(unsigned)((stride0_elems >> 32) & 0xFFFFull);          // dim0_stride[47:32]
  g1[7] = 0;
  const i32x4 z4 = {0, 0, 0, 0};
  const i32x8 z8 = {0, 0, 0, 0, 0, 0, 0, 0};
  __builtin_amdgcn_tensor_load_to_lds(g0, g1, z4, z4, z8, 0);
}

// ---------------------------------------------------------------------------
// bf16 WMMA GEMM:  C[M,N] (f32) = A[M,K] (bf16, row-major) x Wt[N,K]^T (bf16)
// Block tile 128x128, K-step 32. 8 waves: 4 along M x 2 along N; each wave
// owns 2(M) x 4(N) v_wmma_f32_16x16x32_bf16 accumulators.
// Tiles staged via TDM (tensor_load_to_lds), double-buffered on TENSORcnt.
// Uses dynamic LDS only, so LDS byte offsets for the D# start at 0.
// ---------------------------------------------------------------------------
#define GTK   32
#define ROWE  40                    // padded LDS row stride (elements): 80B, 16B-aligned
#define TILE_E   (128 * ROWE)       // 5120 elements = 10240 B per 128x32 tile
#define BUF_E    (2 * TILE_E)       // A-tile + B-tile per buffer
#define BUF_BYTES (BUF_E * 2)       // 20480 B
#define GEMM_LDS_BYTES (2 * BUF_BYTES)

__global__ __launch_bounds__(256) void gemm_bf16_wmma_tdm_kernel(
    const __bf16* __restrict__ A, const __bf16* __restrict__ Wt,
    float* __restrict__ C, int M, int N, int K, int accumulate)
{
  extern __shared__ __bf16 smem[];   // no static LDS in this kernel -> base offset 0

  const int tid   = threadIdx.x;
  const int lane  = tid & 31;
  const int wave  = tid >> 5;
  const int wm    = wave & 3;        // wave tile row (32 rows each)
  const int wn    = wave >> 2;       // wave tile col (64 cols each)
  const int lidx  = lane & 15;
  const int lhalf = lane >> 4;

  const int m0 = blockIdx.y * 128;
  const int n0 = blockIdx.x * 128;

  f32x8 acc[2][4];
  if (accumulate) {
#pragma unroll
    for (int mi = 0; mi < 2; ++mi)
#pragma unroll
      for (int ni = 0; ni < 4; ++ni)
#pragma unroll
        for (int r = 0; r < 8; ++r)
          acc[mi][ni][r] = C[(size_t)(m0 + wm * 32 + mi * 16 + lhalf * 8 + r) * N +
                             (n0 + wn * 64 + ni * 16 + lidx)];
  } else {
#pragma unroll
    for (int mi = 0; mi < 2; ++mi)
#pragma unroll
      for (int ni = 0; ni < 4; ++ni)
#pragma unroll
        for (int r = 0; r < 8; ++r) acc[mi][ni][r] = 0.0f;
  }

  const int NT = K / GTK;
  if (wave == 0) {   // one wave drives the DMA engine (EXEC ignored by TDM)
    tdm_load_2d(0u,               A  + (size_t)m0 * K, (unsigned)K, (unsigned)M,
                GTK, 128, (unsigned long long)K);
    tdm_load_2d((unsigned)(TILE_E * 2), Wt + (size_t)n0 * K, (unsigned)K, (unsigned)N,
                GTK, 128, (unsigned long long)K);
  }

  for (int kt = 0; kt < NT; ++kt) {
    const int cur = kt & 1;
    if (wave == 0) {
      if (kt + 1 < NT) {           // kick next tile's DMA, then wait for current
        const int nxt = cur ^ 1;
        const int k0n = (kt + 1) * GTK;
        tdm_load_2d((unsigned)(nxt * BUF_BYTES),
                    A + (size_t)m0 * K + k0n, (unsigned)K, (unsigned)M,
                    GTK, 128, (unsigned long long)K);
        tdm_load_2d((unsigned)(nxt * BUF_BYTES + TILE_E * 2),
                    Wt + (size_t)n0 * K + k0n, (unsigned)K, (unsigned)N,
                    GTK, 128, (unsigned long long)K);
        __builtin_amdgcn_s_wait_tensorcnt(2);   // current tile's 2 DMAs complete
      } else {
        __builtin_amdgcn_s_wait_tensorcnt(0);
      }
    }
    __syncthreads();               // current buffer visible to all waves

    const __bf16* As = smem + cur * BUF_E;
    const __bf16* Bs = As + TILE_E;

    union Frag { bf16x16 v; uint4 q[2]; };
    Frag a[2], b[4];
    // A fragment, 16-bit 16x32 layout: lanes 0-15 row M, K{0..7,16..23}; lanes 16-31 K{8..15,24..31}
#pragma unroll
    for (int mi = 0; mi < 2; ++mi) {
      const int row = wm * 32 + mi * 16 + lidx;
      a[mi].q[0] = *(const uint4*)(As + row * ROWE + lhalf * 8);
      a[mi].q[1] = *(const uint4*)(As + row * ROWE + 16 + lhalf * 8);
    }
    // B fragment (K x N): lane = column; lanes 0-15 K{0..7,16..23}, lanes 16-31 K{8..15,24..31}
#pragma unroll
    for (int ni = 0; ni < 4; ++ni) {
      const int col = wn * 64 + ni * 16 + lidx;
      b[ni].q[0] = *(const uint4*)(Bs + col * ROWE + lhalf * 8);
      b[ni].q[1] = *(const uint4*)(Bs + col * ROWE + 16 + lhalf * 8);
    }
#pragma unroll
    for (int mi = 0; mi < 2; ++mi)
#pragma unroll
      for (int ni = 0; ni < 4; ++ni)
        acc[mi][ni] = __builtin_amdgcn_wmma_f32_16x16x32_bf16(
            false, a[mi].v, false, b[ni].v, (short)0, acc[mi][ni], false, false);

    __syncthreads();               // all waves done reading before buffer is re-filled
  }

  // D layout: lanes 0-15: M=r, N=lane; lanes 16-31: M=8+r, N=lane-16
#pragma unroll
  for (int mi = 0; mi < 2; ++mi)
#pragma unroll
    for (int ni = 0; ni < 4; ++ni)
#pragma unroll
      for (int r = 0; r < 8; ++r)
        C[(size_t)(m0 + wm * 32 + mi * 16 + lhalf * 8 + r) * N +
          (n0 + wn * 64 + ni * 16 + lidx)] = acc[mi][ni][r];
}

// ---------------------------------------------------------------------------
// conv1 (1x1, 1->C1) + BN + ReLU + SimAM.  One block per (b, c1) plane.
// ---------------------------------------------------------------------------
__global__ __launch_bounds__(256) void conv1_bn_relu_simam_kernel(
    const float* __restrict__ x,  const float* __restrict__ w1, const float* __restrict__ cb,
    const float* __restrict__ g,  const float* __restrict__ bb,
    const float* __restrict__ bm, const float* __restrict__ bv,
    float* __restrict__ y1)
{
  __shared__ float sred[16];
  const int b = blockIdx.x >> 5;   // / C1_
  const int c = blockIdx.x & 31;
  const float inv   = g[c] * rsqrtf(bv[c] + 1e-5f);
  const float scale = w1[c] * inv;
  const float shift = cb[c] * inv + bb[c] - bm[c] * inv;
  const float* xp = x + (size_t)b * (T_ * D_);

  float vals[32];
  float s = 0.f;
#pragma unroll
  for (int j = 0; j < 32; ++j) {
    float v = fmaxf(xp[j * 256 + threadIdx.x] * scale + shift, 0.f);
    vals[j] = v; s += v;
  }
  const float mu = block_reduce_sum(s, sred) * (1.0f / (T_ * D_));
  float sq = 0.f;
#pragma unroll
  for (int j = 0; j < 32; ++j) { const float d = vals[j] - mu; sq += d * d; }
  const float var   = block_reduce_sum(sq, sred) / ((float)(T_ * D_ - 1) + 1e-8f);
  const float denom = 1.0f / (4.0f * (var + 1e-4f) + 1e-8f);

  float* yp = y1 + ((size_t)b * C1_ + c) * (T_ * D_);
#pragma unroll
  for (int j = 0; j < 32; ++j) {
    const float d = vals[j] - mu;
    yp[j * 256 + threadIdx.x] = vals[j] * sigmoidf_(d * d * denom + 0.5f);
  }
}

// ---------------------------------------------------------------------------
// conv2 (3x3, C1->C2, pad 1) + BN + ReLU + SimAM, fused with transpose/pack
// into bf16 seq[t][b][f], f = c2*D + w, t = h.  One block per (b, c2).
// ---------------------------------------------------------------------------
__global__ __launch_bounds__(256) void conv2_bn_relu_simam_pack_kernel(
    const float* __restrict__ y1, const float* __restrict__ w2, const float* __restrict__ cb,
    const float* __restrict__ g,  const float* __restrict__ bb,
    const float* __restrict__ bm, const float* __restrict__ bv,
    __bf16* __restrict__ seq)
{
  __shared__ float wsm[C1_ * 9];
  __shared__ float sred[16];
  const int b  = blockIdx.x >> 6;   // / C2_
  const int c2 = blockIdx.x & 63;
  for (int i = threadIdx.x; i < C1_ * 9; i += 256) wsm[i] = w2[(size_t)c2 * C1_ * 9 + i];
  __syncthreads();

  const float inv   = g[c2] * rsqrtf(bv[c2] + 1e-5f);
  const float shift = cb[c2] * inv + bb[c2] - bm[c2] * inv;
  const float* xin = y1 + (size_t)b * C1_ * T_ * D_;

  float vals[32];
  float s = 0.f;
#pragma unroll
  for (int j = 0; j < 32; ++j) {
    const int idx = j * 256 + threadIdx.x;
    const int h = idx >> 7;    // / D_
    const int w = idx & 127;
    float acc = 0.f;
    for (int c1 = 0; c1 < C1_; ++c1) {
      const float* ip = xin + ((size_t)c1 * T_ + h) * D_ + w;
      const float* wp = &wsm[c1 * 9];
#pragma unroll
      for (int dy = -1; dy <= 1; ++dy) {
        if (h + dy < 0 || h + dy >= T_) continue;
#pragma unroll
        for (int dx = -1; dx <= 1; ++dx) {
          if (w + dx < 0 || w + dx >= D_) continue;
          acc += ip[dy * D_ + dx] * wp[(dy + 1) * 3 + (dx + 1)];
        }
      }
    }
    const float v = fmaxf(acc * inv + shift, 0.f);
    vals[j] = v; s += v;
  }
  const float mu = block_reduce_sum(s, sred) * (1.0f / (T_ * D_));
  float sq = 0.f;
#pragma unroll
  for (int j = 0; j < 32; ++j) { const float d = vals[j] - mu; sq += d * d; }
  const float var   = block_reduce_sum(sq, sred) / ((float)(T_ * D_ - 1) + 1e-8f);
  const float denom = 1.0f / (4.0f * (var + 1e-4f) + 1e-8f);

#pragma unroll
  for (int j = 0; j < 32; ++j) {
    const int idx = j * 256 + threadIdx.x;
    const int h = idx >> 7;
    const int w = idx & 127;
    const float d = vals[j] - mu;
    const float out = vals[j] * sigmoidf_(d * d * denom + 0.5f);
    seq[((size_t)h * B_ + b) * F_ + c2 * D_ + w] = (__bf16)out;
  }
}

// ---------------------------------------------------------------------------
// Small elementwise kernels
// ---------------------------------------------------------------------------
__global__ void castbf_kernel(const float* __restrict__ s, __bf16* __restrict__ d, size_t n) {
  const size_t i = ((size_t)blockIdx.x * blockDim.x + threadIdx.x) * 4;
  if (i + 3 < n) {
    const float4 v = *(const float4*)(s + i);
    d[i + 0] = (__bf16)v.x; d[i + 1] = (__bf16)v.y;
    d[i + 2] = (__bf16)v.z; d[i + 3] = (__bf16)v.w;
  }
}

__global__ void zero2_kernel(float* __restrict__ a, float* __restrict__ b) {
  const size_t i = (size_t)blockIdx.x * blockDim.x + threadIdx.x;
  a[i] = 0.f; b[i] = 0.f;
}

__global__ void initx_kernel(const __bf16* __restrict__ seq_t,
                             float* __restrict__ xm, __bf16* __restrict__ xmb) {
  const size_t i = (size_t)blockIdx.x * blockDim.x + threadIdx.x;
  const __bf16 v = seq_t[i];
  xmb[i] = v; xm[i] = (float)v;
}

__global__ void inith_kernel(const float* __restrict__ h,
                             float* __restrict__ hm, __bf16* __restrict__ hmb) {
  const size_t i = (size_t)blockIdx.x * blockDim.x + threadIdx.x;
  const float v = h[i];
  hm[i] = v; hmb[i] = (__bf16)v;
}

// m = 2*sigmoid(q + bias) * m  (updates both f32 and bf16 copies)
__global__ void mog_scale_kernel(const float* __restrict__ q, const float* __restrict__ bias,
                                 float* __restrict__ m, __bf16* __restrict__ mb, int ncols) {
  const size_t i  = (size_t)blockIdx.x * blockDim.x + threadIdx.x;
  const int   col = (int)(i % (size_t)ncols);
  const float v = 2.0f * sigmoidf_(q[i] + bias[col]) * m[i];
  m[i] = v; mb[i] = (__bf16)v;
}

__global__ void lstm_cell_kernel(const float* __restrict__ gates,
                                 const float* __restrict__ bi, const float* __restrict__ bh,
                                 float* __restrict__ h, float* __restrict__ c) {
  const int i  = blockIdx.x * 256 + threadIdx.x;   // B*H
  const int bb = i >> 10;
  const int j  = i & (H_ - 1);
  const float* gb = gates + (size_t)bb * G4H;
  const float gi = gb[j]          + bi[j]          + bh[j];
  const float gf = gb[H_ + j]     + bi[H_ + j]     + bh[H_ + j];
  const float gg = gb[2 * H_ + j] + bi[2 * H_ + j] + bh[2 * H_ + j];
  const float go = gb[3 * H_ + j] + bi[3 * H_ + j] + bh[3 * H_ + j];
  const float cn = sigmoidf_(gf) * c[i] + sigmoidf_(gi) * tanhf(gg);
  const float hn = sigmoidf_(go) * tanhf(cn);
  c[i] = cn; h[i] = hn;
}

__global__ void fc_kernel(const float* __restrict__ h, const float* __restrict__ w,
                          const float* __restrict__ bias, float* __restrict__ out) {
  const int i = blockIdx.x * blockDim.x + threadIdx.x;
  if (i >= B_ * NC_) return;
  const int bb = i / NC_, n = i % NC_;
  const float* hp = h + (size_t)bb * H_;
  const float* wp = w + (size_t)n * H_;
  float s = 0.f;
  for (int k = 0; k < H_; ++k) s += hp[k] * wp[k];
  out[i] = s + bias[n];
}

// ---------------------------------------------------------------------------
// Host driver
// ---------------------------------------------------------------------------
static inline void run_gemm(const __bf16* A, const __bf16* Wt, float* C,
                            int M, int N, int K, int accum, hipStream_t s) {
  dim3 grid(N / 128, M / 128);
  gemm_bf16_wmma_tdm_kernel<<<grid, 256, GEMM_LDS_BYTES, s>>>(A, Wt, C, M, N, K, accum);
}

extern "C" void kernel_launch(void* const* d_in, const int* in_sizes, int n_in,
                              void* d_out, int out_size, void* d_ws, size_t ws_size,
                              hipStream_t stream) {
  const float* x       = (const float*)d_in[0];
  const float* conv1_w = (const float*)d_in[1];
  const float* conv1_b = (const float*)d_in[2];
  const float* bn1_g   = (const float*)d_in[3];
  const float* bn1_b   = (const float*)d_in[4];
  const float* bn1_m   = (const float*)d_in[5];
  const float* bn1_v   = (const float*)d_in[6];
  const float* conv2_w = (const float*)d_in[7];
  const float* conv2_b = (const float*)d_in[8];
  const float* bn2_g   = (const float*)d_in[9];
  const float* bn2_b   = (const float*)d_in[10];
  const float* bn2_m   = (const float*)d_in[11];
  const float* bn2_v   = (const float*)d_in[12];
  const float* Qw[3]   = {(const float*)d_in[13], (const float*)d_in[15], (const float*)d_in[17]};
  const float* Qb[3]   = {(const float*)d_in[14], (const float*)d_in[16], (const float*)d_in[18]};
  const float* Rw[2]   = {(const float*)d_in[19], (const float*)d_in[21]};
  const float* Rb[2]   = {(const float*)d_in[20], (const float*)d_in[22]};
  const float* W_ih    = (const float*)d_in[23];
  const float* b_ih    = (const float*)d_in[24];
  const float* W_hh    = (const float*)d_in[25];
  const float* b_hh    = (const float*)d_in[26];
  const float* fc_w    = (const float*)d_in[27];
  const float* fc_b    = (const float*)d_in[28];
  float* out = (float*)d_out;

  char* ws = (char*)d_ws;
  size_t off = 0;
  auto walloc = [&](size_t bytes) -> void* {
    void* p = ws + off;
    off = (off + bytes + 255) & ~(size_t)255;
    return p;
  };

  float*  y1    = (float*)walloc((size_t)B_ * C1_ * T_ * D_ * 4);
  __bf16* seq   = (__bf16*)walloc((size_t)T_ * B_ * F_ * 2);
  __bf16* Qwb[3], *Rwb[2];
  for (int i = 0; i < 3; ++i) Qwb[i] = (__bf16*)walloc((size_t)F_ * H_ * 2);
  for (int i = 0; i < 2; ++i) Rwb[i] = (__bf16*)walloc((size_t)H_ * F_ * 2);
  __bf16* Wihb  = (__bf16*)walloc((size_t)G4H * F_ * 2);
  __bf16* Whhb  = (__bf16*)walloc((size_t)G4H * H_ * 2);
  float*  xm    = (float*)walloc((size_t)B_ * F_ * 4);
  __bf16* xmb   = (__bf16*)walloc((size_t)B_ * F_ * 2);
  float*  hm    = (float*)walloc((size_t)B_ * H_ * 4);
  __bf16* hmb   = (__bf16*)walloc((size_t)B_ * H_ * 2);
  float*  hst   = (float*)walloc((size_t)B_ * H_ * 4);
  float*  cst   = (float*)walloc((size_t)B_ * H_ * 4);
  float*  qtmp  = (float*)walloc((size_t)B_ * F_ * 4);
  float*  gates = (float*)walloc((size_t)B_ * G4H * 4);

  auto cast = [&](const float* s, __bf16* d, size_t n) {
    castbf_kernel<<<(unsigned)((n / 4 + 255) / 256), 256, 0, stream>>>(s, d, n);
  };

  // 1) weights -> bf16
  for (int i = 0; i < 3; ++i) cast(Qw[i], Qwb[i], (size_t)F_ * H_);
  for (int i = 0; i < 2; ++i) cast(Rw[i], Rwb[i], (size_t)H_ * F_);
  cast(W_ih, Wihb, (size_t)G4H * F_);
  cast(W_hh, Whhb, (size_t)G4H * H_);

  // 2) conv front-end + pack to seq (bf16, [T][B][F])
  conv1_bn_relu_simam_kernel<<<B_ * C1_, 256, 0, stream>>>(
      x, conv1_w, conv1_b, bn1_g, bn1_b, bn1_m, bn1_v, y1);
  conv2_bn_relu_simam_pack_kernel<<<B_ * C2_, 256, 0, stream>>>(
      y1, conv2_w, conv2_b, bn2_g, bn2_b, bn2_m, bn2_v, seq);

  // 3) LSTM state = 0
  zero2_kernel<<<(B_ * H_) / 256, 256, 0, stream>>>(hst, cst);

  const int gF = (B_ * F_) / 256;
  const int gH = (B_ * H_) / 256;

  // 4) Mogrifier-LSTM over T timesteps
  for (int t = 0; t < T_; ++t) {
    initx_kernel<<<gF, 256, 0, stream>>>(seq + (size_t)t * B_ * F_, xm, xmb);
    inith_kernel<<<gH, 256, 0, stream>>>(hst, hm, hmb);

    // round 1: x = 2*sig(h @ Qw0^T + Qb0) * x
    run_gemm(hmb, Qwb[0], qtmp, B_, F_, H_, 0, stream);
    mog_scale_kernel<<<gF, 256, 0, stream>>>(qtmp, Qb[0], xm, xmb, F_);
    // round 2: h = 2*sig(x @ Rw0^T + Rb0) * h
    run_gemm(xmb, Rwb[0], qtmp, B_, H_, F_, 0, stream);
    mog_scale_kernel<<<gH, 256, 0, stream>>>(qtmp, Rb[0], hm, hmb, H_);
    // round 3
    run_gemm(hmb, Qwb[1], qtmp, B_, F_, H_, 0, stream);
    mog_scale_kernel<<<gF, 256, 0, stream>>>(qtmp, Qb[1], xm, xmb, F_);
    // round 4
    run_gemm(xmb, Rwb[1], qtmp, B_, H_, F_, 0, stream);
    mog_scale_kernel<<<gH, 256, 0, stream>>>(qtmp, Rb[1], hm, hmb, H_);
    // round 5
    run_gemm(hmb, Qwb[2], qtmp, B_, F_, H_, 0, stream);
    mog_scale_kernel<<<gF, 256, 0, stream>>>(qtmp, Qb[2], xm, xmb, F_);

    // gates = x @ W_ih^T + h @ W_hh^T   (biases folded in the cell kernel)
    run_gemm(xmb, Wihb, gates, B_, G4H, F_, 0, stream);
    run_gemm(hmb, Whhb, gates, B_, G4H, H_, 1, stream);

    lstm_cell_kernel<<<gH, 256, 0, stream>>>(gates, b_ih, b_hh, hst, cst);
  }

  // 5) final projection
  fc_kernel<<<(B_ * NC_ + 255) / 256, 256, 0, stream>>>(hst, fc_w, fc_b, out);
  (void)in_sizes; (void)n_in; (void)out_size; (void)ws_size;
}
